// tRNTN_4776003633392
// MI455X (gfx1250) — matrix operand order repaired and checked by reading, not compiled, not run on hardware
//
#include <hip/hip_runtime.h>
#include <hip/hip_bf16.h>

// tRNTN forward for MI455X (gfx1250): wave32, WMMA bf16, async global->LDS staging.
#define VOC   2048
#define D     128
#define RELS  7
#define KT    16384              // kron part of K
#define KL    256                // linear ([l;r]) part of K
#define KTOT  (KT + KL)          // 16640
#define NCH   (KTOT / 32)        // 520 chunks of K=32
#define NSC   (NCH / 4)          // 130 super-chunks of K=128
#define SCB   (4 * 8 * 32 * 16)  // bf16 elements per super-chunk = 16384
#define ROWP  132                // padded LDS row stride (floats)

typedef __bf16 bf16_t;
typedef bf16_t v16bf __attribute__((ext_vector_type(16)));
typedef float  v8f   __attribute__((ext_vector_type(8)));
typedef int    v4i   __attribute__((ext_vector_type(4)));
typedef __attribute__((address_space(1))) v4i* gv4i_p;   // global
typedef __attribute__((address_space(3))) v4i* lv4i_p;   // LDS

#if __has_builtin(__builtin_amdgcn_global_load_async_to_lds_b128) && \
    __has_builtin(__builtin_amdgcn_s_wait_asynccnt)
#define ASYNC_OK 1
#endif

// copy 32 bytes (16 bf16) global -> LDS
__device__ __forceinline__ void cp32(const bf16_t* __restrict__ g, bf16_t* l) {
#ifdef ASYNC_OK
  __builtin_amdgcn_global_load_async_to_lds_b128(
      (gv4i_p)(v4i*)(bf16_t*)g, (lv4i_p)(v4i*)l, 0, 0);
  __builtin_amdgcn_global_load_async_to_lds_b128(
      (gv4i_p)(v4i*)(bf16_t*)(g + 8), (lv4i_p)(v4i*)(l + 8), 0, 0);
#else
  float4 t0 = *(const float4*)g;
  float4 t1 = *(const float4*)(g + 8);
  *(float4*)l = t0;
  *(float4*)(l + 8) = t1;
#endif
}

__device__ __forceinline__ void wait_stage() {
#ifdef ASYNC_OK
  __builtin_amdgcn_s_wait_asynccnt(0);
#endif
  __syncthreads();
}

// all 256 threads copy one 32KB super-chunk of packed weights into an LDS buffer
__device__ __forceinline__ void stage_sc(const bf16_t* __restrict__ Wpk,
                                         bf16_t* dst, int sc, int tid) {
  const bf16_t* g = Wpk + (size_t)sc * SCB + tid * 16;
  bf16_t* l = dst + tid * 16;
#pragma unroll
  for (int r = 0; r < 4; ++r) cp32(g + r * 4096, l + r * 4096);
}

// ---------------------------------------------------------------- embedding
__global__ void embed_kernel(const int* __restrict__ ids,
                             const float* __restrict__ Wv,
                             const float* __restrict__ bv,
                             float* __restrict__ h0) {
  int idx  = blockIdx.x * 256 + threadIdx.x;  // B*2*16*128 total
  int dd   = idx & 127;
  int leaf = idx >> 7;
  int id   = ids[leaf];
  h0[idx]  = Wv[dd * VOC + id] + bv[dd];
}

// ------------------------------------------------- weight prepack -> bf16 B-frags
// Layout: [chunk c][o-tile][lane n][e], value = W[o = ot*16+(n&15)][k],
// k = c*32 + (n>=16 ? 16 : 0) + e.  k<KT -> Wt (o x 16384), else Wl (o x 256).
__global__ void prepack_kernel(const float* __restrict__ Wt,
                               const float* __restrict__ Wl,
                               bf16_t* __restrict__ out) {
  int idx = blockIdx.x * 256 + threadIdx.x;   // 520*8*32*16 = 2,129,920 exact
  int e   = idx & 15;
  int n   = (idx >> 4) & 31;
  int ot  = (idx >> 9) & 7;
  int c   = idx >> 12;
  int o   = ot * 16 + (n & 15);
  int k   = c * 32 + ((n >> 4) << 4) + e;
  float v = (k < KT) ? Wt[(size_t)o * KT + k] : Wl[o * KL + (k - KT)];
  out[idx] = (bf16_t)v;
}

// ---------------------------------------------------------------- core GEMM
// One wave = 16 output nodes x 128 outputs, K = 16640.
// B (packed bf16 weights) streamed through a double-buffered LDS stage shared by
// all 8 waves of the block; A (16x32 kron tile) built on the fly from LDS l/r.
__device__ __forceinline__ void tile_gemm(const float* __restrict__ hin,
                                          const bf16_t* __restrict__ Wpk,
                                          float* Lsh, float* Rsh, bf16_t* Bsh,
                                          int nodeBase, int shift,
                                          int lane, int tid, v8f acc[8]) {
  const int hi   = lane >> 4;
  const int mrow = lane & 15;
  const int nP   = 1 << shift;

  // stage l/r rows of this wave's 16 nodes into LDS (padded rows)
  for (int m = 0; m < 16; ++m) {
    int g  = nodeBase + m;
    int bs = g >> shift;
    int p  = g & (nP - 1);
    size_t lidx = (size_t)(bs * 2 * nP + 2 * p) * D;
    float4 lv = *(const float4*)(hin + lidx + lane * 4);
    float4 rv = *(const float4*)(hin + lidx + D + lane * 4);
    *(float4*)(Lsh + m * ROWP + lane * 4) = lv;
    *(float4*)(Rsh + m * ROWP + lane * 4) = rv;
  }

  // prime first super-chunk
  stage_sc(Wpk, Bsh, 0, tid);

  for (int sc = 0; sc < NSC; ++sc) {
    wait_stage();                       // chunk sc ready; all waves done with sc-1
    if (sc + 1 < NSC) stage_sc(Wpk, Bsh + ((sc + 1) & 1) * SCB, sc + 1, tid);

    // kron super-chunks: A[m, k] = l[m,sc] * r[m, j]; tail: A[m,k] = [l;r][t]
    const float* src  = (sc == 128) ? Lsh : Rsh;
    const float scale = (sc < 128) ? Lsh[mrow * ROWP + sc] : 1.0f;
    const float*  arow = src + mrow * ROWP + hi * 16;
    const bf16_t* bb   = Bsh + (sc & 1) * SCB + lane * 16;

#pragma unroll
    for (int q = 0; q < 4; ++q) {       // chunk c = sc*4 + q, j0 = q*32 + hi*16
      const float* rp = arow + q * 32;
      float4 p0 = *(const float4*)(rp);
      float4 p1 = *(const float4*)(rp + 4);
      float4 q0 = *(const float4*)(rp + 8);
      float4 q1 = *(const float4*)(rp + 12);

      v16bf a;
      a[0]  = (bf16_t)(scale * p0.x);  a[1]  = (bf16_t)(scale * p0.y);
      a[2]  = (bf16_t)(scale * p0.z);  a[3]  = (bf16_t)(scale * p0.w);
      a[4]  = (bf16_t)(scale * p1.x);  a[5]  = (bf16_t)(scale * p1.y);
      a[6]  = (bf16_t)(scale * p1.z);  a[7]  = (bf16_t)(scale * p1.w);
      a[8]  = (bf16_t)(scale * q0.x);  a[9]  = (bf16_t)(scale * q0.y);
      a[10] = (bf16_t)(scale * q0.z);  a[11] = (bf16_t)(scale * q0.w);
      a[12] = (bf16_t)(scale * q1.x);  a[13] = (bf16_t)(scale * q1.y);
      a[14] = (bf16_t)(scale * q1.z);  a[15] = (bf16_t)(scale * q1.w);

#pragma unroll
      for (int ot = 0; ot < 8; ++ot) {
        v16bf bfr = *(const v16bf*)(bb + q * 4096 + ot * (32 * 16));
        acc[ot] = __builtin_amdgcn_wmma_f32_16x16x32_bf16(
            false, a, false, bfr, (short)0, acc[ot], false, false);
      }
    }
  }
}

// ------------------------------------------------------------- combine level
__global__ __launch_bounds__(256, 1)
void combine_kernel(const float* __restrict__ hin, float* __restrict__ hout,
                    const bf16_t* __restrict__ Wpk,
                    const float* __restrict__ b1, const float* __restrict__ b2,
                    int shift) {
  __shared__ __align__(16) float  sh[8][2][16][ROWP];
  __shared__ __align__(16) bf16_t bsh[2][SCB];
  const int lane = threadIdx.x & 31;
  const int w    = threadIdx.x >> 5;
  const int hi   = lane >> 4;
  const int mrow = lane & 15;
  const int nodeBase = (blockIdx.x * 8 + w) * 16;

  v8f acc[8] = {};
  tile_gemm(hin, Wpk, &sh[w][0][0][0], &sh[w][1][0][0], &bsh[0][0],
            nodeBase, shift, lane, threadIdx.x, acc);

#pragma unroll
  for (int ot = 0; ot < 8; ++ot) {
    int o = ot * 16 + mrow;
    float bsum = b1[o] + b2[o];
#pragma unroll
    for (int g = 0; g < 8; ++g) {
      int node = nodeBase + g + hi * 8;      // C/D layout: M = g + 8*hi, N = mrow
      hout[(size_t)node * D + o] = tanhf(acc[ot][g] + bsum);
    }
  }
}

// ------------------------------------------------- final combine + log_softmax
__global__ __launch_bounds__(256, 1)
void final_kernel(const float* __restrict__ hin, float* __restrict__ out,
                  const bf16_t* __restrict__ Wpk,
                  const float* __restrict__ bcpr, const float* __restrict__ bcprt,
                  const float* __restrict__ Wsm, const float* __restrict__ bsm) {
  __shared__ __align__(16) float  sh[8][2][16][ROWP];
  __shared__ __align__(16) bf16_t bsh[2][SCB];
  const int lane = threadIdx.x & 31;
  const int w    = threadIdx.x >> 5;
  const int hi   = lane >> 4;
  const int mrow = lane & 15;
  const int nodeBase = (blockIdx.x * 8 + w) * 16;

  v8f acc[8] = {};
  tile_gemm(hin, Wpk, &sh[w][0][0][0], &sh[w][1][0][0], &bsh[0][0],
            nodeBase, /*shift=*/0, lane, threadIdx.x, acc);

  // leaky-ReLU activations into LDS (reuse L staging region, GEMM is done)
  __syncthreads();
  float* Act = &sh[w][0][0][0];
#pragma unroll
  for (int ot = 0; ot < 8; ++ot) {
    int o = ot * 16 + mrow;
    float bsum = bcpr[o] + bcprt[o];
#pragma unroll
    for (int g = 0; g < 8; ++g) {
      int m = g + hi * 8;
      float v = acc[ot][g] + bsum;
      Act[m * ROWP + o] = v > 0.0f ? v : 0.01f * v;
    }
  }
  __syncthreads();

  if (lane < 16) {
    int node = nodeBase + lane;
    const float* av = Act + lane * ROWP;
    float lg[RELS];
    float mx = -1e30f;
#pragma unroll
    for (int rel = 0; rel < RELS; ++rel) {
      float s = bsm[rel];
      for (int o = 0; o < D; ++o) s += Wsm[rel * D + o] * av[o];
      lg[rel] = s;
      mx = fmaxf(mx, s);
    }
    float se = 0.0f;
#pragma unroll
    for (int rel = 0; rel < RELS; ++rel) se += expf(lg[rel] - mx);
    float lse = mx + logf(se);
#pragma unroll
    for (int rel = 0; rel < RELS; ++rel) out[node * RELS + rel] = lg[rel] - lse;
  }
}

// ---------------------------------------------------------------- launcher
extern "C" void kernel_launch(void* const* d_in, const int* in_sizes, int n_in,
                              void* d_out, int out_size, void* d_ws, size_t ws_size,
                              hipStream_t stream) {
  (void)in_sizes; (void)n_in; (void)out_size; (void)ws_size;
  const int*   ids   = (const int*)  d_in[0];
  const float* Wv    = (const float*)d_in[1];
  const float* bv    = (const float*)d_in[2];
  const float* Wcps  = (const float*)d_in[3];
  const float* bcps  = (const float*)d_in[4];
  const float* Wcpst = (const float*)d_in[5];
  const float* bcpst = (const float*)d_in[6];
  const float* Wcpr  = (const float*)d_in[7];
  const float* bcpr  = (const float*)d_in[8];
  const float* Wcprt = (const float*)d_in[9];
  const float* bcprt = (const float*)d_in[10];
  const float* Wsm   = (const float*)d_in[11];
  const float* bsm   = (const float*)d_in[12];
  float* out = (float*)d_out;

  char* ws = (char*)d_ws;
  float*  h0   = (float*)(ws);                                  // 8 MB ping
  float*  h1   = (float*)(ws + (size_t)(8u << 20));             // 4 MB pong
  bf16_t* Wpk1 = (bf16_t*)(ws + (size_t)(12u << 20));           // 4.06 MB
  bf16_t* Wpk2 = (bf16_t*)(ws + (size_t)(12u << 20) + 4260096); // 4.06 MB

  prepack_kernel<<<8320, 256, 0, stream>>>(Wcpst, Wcps, Wpk1);
  prepack_kernel<<<8320, 256, 0, stream>>>(Wcprt, Wcpr, Wpk2);
  embed_kernel  <<<8192, 256, 0, stream>>>(ids, Wv, bv, h0);

  // tree levels: 16->8->4->2->1 leaves per side (ping-pong h0/h1)
  combine_kernel<<<64, 256, 0, stream>>>(h0, h1, Wpk1, bcps, bcpst, 3);
  combine_kernel<<<32, 256, 0, stream>>>(h1, h0, Wpk1, bcps, bcpst, 2);
  combine_kernel<<<16, 256, 0, stream>>>(h0, h1, Wpk1, bcps, bcpst, 1);
  combine_kernel<<< 8, 256, 0, stream>>>(h1, h0, Wpk1, bcps, bcpst, 0);

  final_kernel<<<4, 256, 0, stream>>>(h0, out, Wpk2, bcpr, bcprt, Wsm, bsm);
}
